// EfficientAttentionBlock_65515431133665
// MI455X (gfx1250) — matrix operand order repaired and checked by reading, not compiled, not run on hardware
//
#include <hip/hip_runtime.h>

// ---------------- types ----------------
typedef __bf16 bf16_t;
typedef bf16_t v16bf __attribute__((ext_vector_type(16)));
typedef float  v8f   __attribute__((ext_vector_type(8)));

#define WSZ 8
#define CIN 256
#define IMG 256

// LDS strides (in elements)
#define XS_STRIDE 272   // X[p=64][c=256] bf16 (padded)
#define VS_STRIDE 68    // V[c=256][q=64] bf16 (union with X region: 64*272 == 256*68)
#define QKS_STRIDE 68   // QK^T [p=64][row=64] bf16
#define SS_STRIDE 68    // S [p=64][q=64] f32; attn bf16 aliased row-wise

union FragU { v16bf v; unsigned int d[8]; };

// native f32 -> bf16 (RNE) via hardware convert
__device__ __forceinline__ unsigned short f2bf(float f) {
  return __builtin_bit_cast(unsigned short, static_cast<__bf16>(f));
}
__device__ __forceinline__ unsigned int f2bf2(float lo, float hi) {
  return (unsigned int)f2bf(lo) | ((unsigned int)f2bf(hi) << 16);
}

__device__ __forceinline__ v8f wmma_bf16(v16bf a, v16bf b, v8f acc) {
  // D = A(16x32) * B(32x16) + C, f32 accumulate
  return __builtin_amdgcn_wmma_f32_16x16x32_bf16(false, a, false, b, (short)0, acc,
                                                 false, false);
}

// A fragment 16x32 bf16 from row-major [M][K] source (stride in u16 units).
__device__ __forceinline__ v16bf load_a_frag(const unsigned short* base, int stride,
                                             int mrow, int kbase, int lane) {
  const int r = mrow + (lane & 15);
  const int g = (lane >> 4) & 1;
  const unsigned short* p = base + r * stride + kbase + g * 8;
  FragU f;
#pragma unroll
  for (int j = 0; j < 4; ++j) {
    f.d[j]     = *reinterpret_cast<const unsigned int*>(p + 2 * j);
    f.d[4 + j] = *reinterpret_cast<const unsigned int*>(p + 16 + 2 * j);
  }
  return f.v;
}

// B fragment 32x16 bf16 (KxN) from a source stored transposed [N][K] row-major.
__device__ __forceinline__ v16bf load_b_frag(const unsigned short* base, int stride,
                                             int ncol, int kbase, int lane) {
  const int n = ncol + (lane & 15);
  const int k = kbase + ((lane >> 4) & 1) * 8;
  const unsigned short* p = base + n * stride + k;
  FragU f;
#pragma unroll
  for (int j = 0; j < 4; ++j) {
    f.d[j]     = *reinterpret_cast<const unsigned int*>(p + 2 * j);
    f.d[4 + j] = *reinterpret_cast<const unsigned int*>(p + 16 + 2 * j);
  }
  return f.v;
}

// ---------------- prepass: fp32 weights -> bf16 in workspace ----------------
__global__ void cvt_weights_kernel(const float* __restrict__ wq,
                                   const float* __restrict__ wk,
                                   const float* __restrict__ wv,
                                   unsigned short* __restrict__ wqk_bf,
                                   unsigned short* __restrict__ wv_bf) {
  int i = blockIdx.x * blockDim.x + threadIdx.x;
  if (i < 64 * 256) {  // rows 0..31 = wq, 32..63 = wk
    int r = i >> 8;
    float v = (r < 32) ? wq[i] : wk[i - 32 * 256];
    wqk_bf[i] = f2bf(v);
  }
  if (i < 256 * 256) wv_bf[i] = f2bf(wv[i]);
}

// ---------------- main kernel: one workgroup per 8x8 window ----------------
__global__ __launch_bounds__(256)
void winattn_kernel(const float* __restrict__ x, const float* __restrict__ bq,
                    const float* __restrict__ bk, const float* __restrict__ bv,
                    const float* __restrict__ gamma_p,
                    const unsigned short* __restrict__ wqk_bf,
                    const unsigned short* __restrict__ wv_bf,
                    float* __restrict__ out) {
  __shared__ __align__(16) unsigned short lds_xv[64 * XS_STRIDE];  // X[p][c] then V[c][q]
  __shared__ __align__(16) unsigned short lds_qk[64 * QKS_STRIDE]; // [p][Q rows 0-31 | K rows 32-63]
  __shared__ __align__(16) float          lds_s[64 * SS_STRIDE];   // scores f32 / attn bf16 alias

  const int tid  = threadIdx.x;
  const int lane = tid & 31;
  const int wid  = tid >> 5;            // wave 0..7
  const int g    = (lane >> 4) & 1;
  const int l15  = lane & 15;

  const int n  = blockIdx.x;            // window id
  const int b_ = n >> 10;
  const int hw = (n >> 5) & 31;
  const int ww = n & 31;
  const size_t img_base = (size_t)b_ * CIN * IMG * IMG;
  const int y0 = hw * WSZ, x0 = ww * WSZ;

  // per-wave tiling for the 64-row GEMMs (QK, S): A-fragment shared by both tiles
  const int mtq = wid >> 1;             // p/row tile 0..3
  const int ntb = (wid & 1) * 2;        // first of two n tiles

  // ---- stage X window -> LDS bf16, transposed [p][c] (float4 loads, coalesced 32B)
#pragma unroll
  for (int it = 0; it < 16; ++it) {
    int i   = tid + it * 256;           // (c, py, px4)
    int px4 = (i & 1) * 4;
    int py  = (i >> 1) & 7;
    int c   = i >> 4;
    const float4 v = *reinterpret_cast<const float4*>(
        x + img_base + ((size_t)c * IMG + (y0 + py)) * IMG + x0 + px4);
    int p = py * WSZ + px4;
    lds_xv[(p + 0) * XS_STRIDE + c] = f2bf(v.x);
    lds_xv[(p + 1) * XS_STRIDE + c] = f2bf(v.y);
    lds_xv[(p + 2) * XS_STRIDE + c] = f2bf(v.z);
    lds_xv[(p + 3) * XS_STRIDE + c] = f2bf(v.w);
  }
  __syncthreads();

  // ---- fused Q|K projection: [64][64] = Wqk[64][256] x X[256][64]
  {
    v8f acc0 = {}, acc1 = {};
#pragma unroll
    for (int ks = 0; ks < 8; ++ks) {
      v16bf a  = load_a_frag(wqk_bf, 256, mtq * 16, ks * 32, lane);
      v16bf b0 = load_b_frag(lds_xv, XS_STRIDE, ntb * 16, ks * 32, lane);
      v16bf b1 = load_b_frag(lds_xv, XS_STRIDE, (ntb + 1) * 16, ks * 32, lane);
      acc0 = wmma_bf16(a, b0, acc0);
      acc1 = wmma_bf16(a, b1, acc1);
    }
    // bias + transposed store (packed dword stores: consecutive rows per lane)
    const int rb = mtq * 16 + 8 * g;                       // first of 8 rows
    const float* bias = (mtq < 2) ? (bq + rb) : (bk + rb - 32);
#pragma unroll
    for (int t = 0; t < 2; ++t) {
      const v8f acc = t ? acc1 : acc0;
      const int pp  = (ntb + t) * 16 + l15;
      unsigned int* dst = reinterpret_cast<unsigned int*>(lds_qk + pp * QKS_STRIDE + rb);
#pragma unroll
      for (int jj = 0; jj < 4; ++jj)
        dst[jj] = f2bf2(acc[2 * jj] + bias[2 * jj], acc[2 * jj + 1] + bias[2 * jj + 1]);
    }
  }

  // ---- V projection into registers: wave covers c in [wid*32, wid*32+32)
  const v8f vzero = {};
  v8f vacc[8];
#pragma unroll
  for (int i = 0; i < 8; ++i) vacc[i] = vzero;
#pragma unroll
  for (int ks = 0; ks < 8; ++ks) {
    v16bf a0 = load_a_frag(wv_bf, 256, (wid * 2 + 0) * 16, ks * 32, lane);
    v16bf a1 = load_a_frag(wv_bf, 256, (wid * 2 + 1) * 16, ks * 32, lane);
#pragma unroll
    for (int nt = 0; nt < 4; ++nt) {
      v16bf bb = load_b_frag(lds_xv, XS_STRIDE, nt * 16, ks * 32, lane);
      vacc[nt]     = wmma_bf16(a0, bb, vacc[nt]);
      vacc[4 + nt] = wmma_bf16(a1, bb, vacc[4 + nt]);
    }
  }
  __syncthreads();    // X reads done; QK^T complete

  // ---- spill V (+bias) as bf16 over the X region: V[c][q]
#pragma unroll
  for (int mi = 0; mi < 2; ++mi) {
#pragma unroll
    for (int nt = 0; nt < 4; ++nt) {
      const int q = nt * 16 + l15;
#pragma unroll
      for (int j = 0; j < 8; ++j) {
        int c = (wid * 2 + mi) * 16 + j + 8 * g;
        lds_xv[c * VS_STRIDE + q] = f2bf(vacc[mi * 4 + nt][j] + bv[c]);
      }
    }
  }

  // ---- scores S[p][q] = sum_o Q[o][p] K[o][q]  (single K=32 WMMA per tile)
  {
    v16bf a  = load_a_frag(lds_qk, QKS_STRIDE, mtq * 16, 0, lane);   // Q^T (cols 0..31)
    v16bf b0 = load_b_frag(lds_qk, QKS_STRIDE, ntb * 16, 32, lane);  // K^T (cols 32..63)
    v16bf b1 = load_b_frag(lds_qk, QKS_STRIDE, (ntb + 1) * 16, 32, lane);
    v8f s0 = {}, s1 = {};
    s0 = wmma_bf16(a, b0, s0);
    s1 = wmma_bf16(a, b1, s1);
#pragma unroll
    for (int j = 0; j < 8; ++j) {
      const int p = mtq * 16 + j + 8 * g;
      lds_s[p * SS_STRIDE + ntb * 16 + l15]       = s0[j];
      lds_s[p * SS_STRIDE + (ntb + 1) * 16 + l15] = s1[j];
    }
  }
  __syncthreads();

  // ---- softmax over q (scale 1/sqrt(64) folded into exp), attn bf16 in place
  if (tid < 64) {
    float rv[64];
#pragma unroll
    for (int i = 0; i < 64; ++i) rv[i] = lds_s[tid * SS_STRIDE + i];
    float mx = rv[0];
#pragma unroll
    for (int i = 1; i < 64; ++i) mx = fmaxf(mx, rv[i]);
    float sum = 0.f;
#pragma unroll
    for (int i = 0; i < 64; ++i) { rv[i] = __expf((rv[i] - mx) * 0.125f); sum += rv[i]; }
    const float inv = 1.0f / sum;
    // packed 16B stores: row start = tid*272 bytes (16-aligned), 17 uint4 per row
    uint4* ar = reinterpret_cast<uint4*>(lds_s) + tid * 17;
#pragma unroll
    for (int i = 0; i < 8; ++i) {
      uint4 w;
      w.x = f2bf2(rv[8 * i + 0] * inv, rv[8 * i + 1] * inv);
      w.y = f2bf2(rv[8 * i + 2] * inv, rv[8 * i + 3] * inv);
      w.z = f2bf2(rv[8 * i + 4] * inv, rv[8 * i + 5] * inv);
      w.w = f2bf2(rv[8 * i + 6] * inv, rv[8 * i + 7] * inv);
      ar[i] = w;
    }
  }
  __syncthreads();

  // ---- O[c][p] = sum_q V[c][q] * attn[p][q]; epilogue gamma*O + x
  const float gm = gamma_p[0];
  const unsigned short* attn = reinterpret_cast<const unsigned short*>(lds_s);
  // hoist the 4 V A-fragments (shared across all 4 p-tiles)
  v16bf av[4];
#pragma unroll
  for (int mi = 0; mi < 2; ++mi)
#pragma unroll
    for (int ks = 0; ks < 2; ++ks)
      av[mi * 2 + ks] = load_a_frag(lds_xv, VS_STRIDE, (wid * 2 + mi) * 16, ks * 32, lane);

#pragma unroll
  for (int nt = 0; nt < 4; ++nt) {
    const v16bf b0 = load_b_frag(attn, SS_STRIDE * 2, nt * 16, 0, lane);
    const v16bf b1 = load_b_frag(attn, SS_STRIDE * 2, nt * 16, 32, lane);
    const int p  = nt * 16 + l15;
    const int py = p >> 3, px = p & 7;
#pragma unroll
    for (int mi = 0; mi < 2; ++mi) {
      v8f oacc = {};
      oacc = wmma_bf16(av[mi * 2 + 0], b0, oacc);
      oacc = wmma_bf16(av[mi * 2 + 1], b1, oacc);
#pragma unroll
      for (int j = 0; j < 8; ++j) {
        int c = (wid * 2 + mi) * 16 + j + 8 * g;
        size_t idx = img_base + ((size_t)c * IMG + (y0 + py)) * IMG + x0 + px;
        out[idx] = gm * oacc[j] + x[idx];
      }
    }
  }
}

extern "C" void kernel_launch(void* const* d_in, const int* in_sizes, int n_in,
                              void* d_out, int out_size, void* d_ws, size_t ws_size,
                              hipStream_t stream) {
  (void)in_sizes; (void)n_in; (void)out_size; (void)ws_size;
  const float* x     = (const float*)d_in[0];
  const float* wq    = (const float*)d_in[1];
  const float* bq    = (const float*)d_in[2];
  const float* wk    = (const float*)d_in[3];
  const float* bk    = (const float*)d_in[4];
  const float* wv    = (const float*)d_in[5];
  const float* bv    = (const float*)d_in[6];
  const float* gamma = (const float*)d_in[7];
  float* out = (float*)d_out;

  unsigned short* wqk_bf = (unsigned short*)d_ws;            // 64*256 bf16
  unsigned short* wv_bf  = wqk_bf + 64 * 256;                // 256*256 bf16

  cvt_weights_kernel<<<256, 256, 0, stream>>>(wq, wk, wv, wqk_bf, wv_bf);
  winattn_kernel<<<4096, 256, 0, stream>>>(x, bq, bk, bv, gamma, wqk_bf, wv_bf, out);
}
